// PPNet_40750649705211
// MI455X (gfx1250) — matrix operand order repaired and checked by reading, not compiled, not run on hardware
//
#include <hip/hip_runtime.h>
#include <hip/hip_bf16.h>

typedef _Float16 half8  __attribute__((ext_vector_type(8)));
typedef _Float16 half16 __attribute__((ext_vector_type(16)));
typedef float    float8 __attribute__((ext_vector_type(8)));
typedef float    f32x4  __attribute__((ext_vector_type(4)));

#define BIG_INF 3.0e38f

// ---------------------------------------------------------------------------
// fp32 -> fp16 cast (weights)
// ---------------------------------------------------------------------------
__global__ __launch_bounds__(256)
void cast_f16_kernel(const float* __restrict__ in, _Float16* __restrict__ out, int n) {
    int i = blockIdx.x * 256 + threadIdx.x;
    if (i < n) out[i] = (_Float16)in[i];
}

// ---------------------------------------------------------------------------
// im2col for the stride-16 16x16 patch conv: P[row][k], row=(b*14+ph)*14+pw,
// k = c*256 + kh*16 + kw  (matches OIHW weight flattening)
// ---------------------------------------------------------------------------
__global__ __launch_bounds__(256)
void patchify_kernel(const float* __restrict__ x, _Float16* __restrict__ P) {
    int row = blockIdx.x;                 // 0..3135
    int pw  = row % 14;
    int t   = row / 14;
    int ph  = t % 14;
    int b   = t / 14;
    for (int k = threadIdx.x; k < 768; k += 256) {
        int c   = k >> 8;
        int rem = k & 255;
        int kh  = rem >> 4;
        int kw  = rem & 15;
        float v = x[(((size_t)(b * 3 + c)) * 224 + (ph * 16 + kh)) * 224 + (pw * 16 + kw)];
        P[(size_t)row * 768 + k] = (_Float16)v;
    }
}

// ---------------------------------------------------------------------------
// WMMA GEMM: C[M,N] = epi( A[M,K] * W[N,K]^T + bias[N] )
// One wave computes a 16x32 output tile (two 16x16 WMMA tiles sharing the A
// fragment -> half the A traffic, 2 WMMAs per 6 b128 loads). 4 waves/block.
// A-tile: lane r=lane&15 owns row M=r; hf=lane>>4 selects K-slice
//   halves j=0..7  : K = k0 + 16*hf + j         (one b128 load)
//   halves j=8..15 : K = k0 + 16*hf + j         (one b128 load)
// B-tile (= W^T, 32x16): lane r owns column N=r; halves j=0..15 : K = k0+16*hf+j
// C/D f32 layout: VGPR v, lane -> element (M = 8*hf + v, N = r)
// epi: 0 = ReLU, 1 = sigmoid
// ---------------------------------------------------------------------------
__global__ __launch_bounds__(128)
void gemm_wmma_kernel(const _Float16* __restrict__ A, const _Float16* __restrict__ W,
                      const float* __restrict__ bias, _Float16* __restrict__ C,
                      float* __restrict__ Cf32, int M, int N, int K, int epi) {
    const int wave    = threadIdx.x >> 5;
    const int lane    = threadIdx.x & 31;
    const int tilesN2 = N >> 5;                  // pairs of N-tiles
    const int nTiles  = (M >> 4) * tilesN2;
    const int tile    = blockIdx.x * 4 + wave;
    if (tile >= nTiles) return;                  // uniform per-wave: EXEC all-1s inside

    const int tm = tile / tilesN2;
    const int tn = tile % tilesN2;               // selects a 32-wide N slab
    const int r  = lane & 15;
    const int hf = lane >> 4;

    const _Float16* Arow  = A + (size_t)(tm * 16 + r) * K;
    const _Float16* Wrow0 = W + (size_t)(tn * 32 + r) * K;        // N tile 0
    const _Float16* Wrow1 = W + (size_t)(tn * 32 + 16 + r) * K;   // N tile 1

    float8 acc0 = {}, acc1 = {};
    for (int k0 = 0; k0 < K; k0 += 32) {
        __builtin_prefetch(Arow + k0 + 64, 0, 0);   // global_prefetch_b8
        half8 a_lo  = *(const half8*)(Arow  + k0 + hf * 16);
        half8 a_hi  = *(const half8*)(Arow  + k0 + hf * 16 + 8);
        half8 b0_lo = *(const half8*)(Wrow0 + k0 + hf * 16);
        half8 b0_hi = *(const half8*)(Wrow0 + k0 + hf * 16 + 8);
        half8 b1_lo = *(const half8*)(Wrow1 + k0 + hf * 16);
        half8 b1_hi = *(const half8*)(Wrow1 + k0 + hf * 16 + 8);
        half16 a, b0, b1;
#pragma unroll
        for (int i = 0; i < 8; ++i) {
            a[i]  = a_lo[i];  a[i + 8]  = a_hi[i];
            b0[i] = b0_lo[i]; b0[i + 8] = b0_hi[i];
            b1[i] = b1_lo[i]; b1[i + 8] = b1_hi[i];
        }
        acc0 = __builtin_amdgcn_wmma_f32_16x16x32_f16(
            false, a, false, b0, (short)0, acc0, false, false);
        acc1 = __builtin_amdgcn_wmma_f32_16x16x32_f16(
            false, a, false, b1, (short)0, acc1, false, false);
    }

    const int   n0  = tn * 32 + r;
    const int   n1  = n0 + 16;
    const float bn0 = bias ? bias[n0] : 0.0f;
    const float bn1 = bias ? bias[n1] : 0.0f;
#pragma unroll
    for (int v = 0; v < 8; ++v) {
        int   m    = tm * 16 + hf * 8 + v;
        float val0 = acc0[v] + bn0;
        float val1 = acc1[v] + bn1;
        if (epi == 0) { val0 = fmaxf(val0, 0.0f);           val1 = fmaxf(val1, 0.0f); }
        else          { val0 = 1.0f / (1.0f + expf(-val0)); val1 = 1.0f / (1.0f + expf(-val1)); }
        C[(size_t)m * N + n0] = (_Float16)val0;
        C[(size_t)m * N + n1] = (_Float16)val1;
        if (Cf32) {
            Cf32[(size_t)m * N + n0] = val0;
            Cf32[(size_t)m * N + n1] = val1;
        }
    }
}

// ---------------------------------------------------------------------------
// Per-(batch, prototype): L2 distances over the 14x14 grid, act map, and the
// two top-5 means (5-smallest distance, 5-largest activation).
// ---------------------------------------------------------------------------
__global__ __launch_bounds__(256)
void dist_topk_kernel(const float* __restrict__ H4, const float* __restrict__ protos,
                      const int* __restrict__ topk_ptr, float* __restrict__ acts,
                      float* __restrict__ proto_act, float* __restrict__ min_d) {
    __shared__ float proto[128];
    __shared__ float dists[256];
    __shared__ float actv[256];
    __shared__ float sval[256];
    __shared__ int   sidx[256];
    __shared__ float accum;

    const int bp = blockIdx.x;            // 0..959
    const int b  = bp / 60;
    const int p  = bp % 60;
    const int t  = threadIdx.x;

    if (t < 128) proto[t] = protos[p * 128 + t];
    __syncthreads();

    float dval = BIG_INF, aval = -BIG_INF;
    if (t < 196) {
        const float* hv = H4 + (size_t)(b * 196 + t) * 128;
        float inner = 0.f, f2 = 0.f, p2 = 0.f;
        for (int k = 0; k < 128; ++k) {
            float hh = hv[k], pp = proto[k];
            inner += hh * pp;
            f2    += hh * hh;
            p2    += pp * pp;
        }
        float d = sqrtf(fmaxf(f2 + p2 - 2.0f * inner, 0.0f));
        dval = d;
        aval = logf((d + 1.0f) / (d + 1e-4f));
        acts[(size_t)bp * 196 + t] = aval;
    }
    dists[t] = dval;
    actv[t]  = aval;
    __syncthreads();

    int K = topk_ptr ? *topk_ptr : 5;
    if (K < 1)   K = 1;
    if (K > 196) K = 196;

    // top-K smallest distance
    if (t == 0) accum = 0.0f;
    __syncthreads();
    for (int it = 0; it < K; ++it) {
        sval[t] = dists[t]; sidx[t] = t;
        __syncthreads();
        for (int s = 128; s > 0; s >>= 1) {
            if (t < s && sval[t + s] < sval[t]) { sval[t] = sval[t + s]; sidx[t] = sidx[t + s]; }
            __syncthreads();
        }
        if (t == 0) { accum += sval[0]; dists[sidx[0]] = BIG_INF; }
        __syncthreads();
    }
    if (t == 0) min_d[bp] = accum / (float)K;

    // top-K largest activation
    if (t == 0) accum = 0.0f;
    __syncthreads();
    for (int it = 0; it < K; ++it) {
        sval[t] = actv[t]; sidx[t] = t;
        __syncthreads();
        for (int s = 128; s > 0; s >>= 1) {
            if (t < s && sval[t + s] > sval[t]) { sval[t] = sval[t + s]; sidx[t] = sidx[t + s]; }
            __syncthreads();
        }
        if (t == 0) { accum += sval[0]; actv[sidx[0]] = -BIG_INF; }
        __syncthreads();
    }
    if (t == 0) proto_act[bp] = accum / (float)K;
}

// ---------------------------------------------------------------------------
// logits = proto_act(16x60) @ last_W(3x60)^T
// ---------------------------------------------------------------------------
__global__ __launch_bounds__(64)
void logits_kernel(const float* __restrict__ pa, const float* __restrict__ lastW,
                   float* __restrict__ out) {
    int t = threadIdx.x;
    if (t >= 48) return;
    int b = t / 3, c = t % 3;
    float s = 0.f;
    for (int p = 0; p < 60; ++p) s += pa[b * 60 + p] * lastW[c * 60 + p];
    out[b * 3 + c] = s;
}

// ---------------------------------------------------------------------------
// 16x nearest-neighbor upsample of the 14x14 act maps to 224x224.
// float4 NON-TEMPORAL stores: the 193 MB output is write-once and ~= the
// 192 MB L2, so NT keeps the hot acts/weights resident instead of being
// evicted by the stream. Each float4 lies within one 16-wide repeat group
// (w0 % 4 == 0 => w0 % 16 in {0,4,8,12}). Fully coalesced b128 stores.
// ---------------------------------------------------------------------------
__global__ __launch_bounds__(256)
void upsample_kernel(const float* __restrict__ acts, f32x4* __restrict__ out, int total4) {
    int i = blockIdx.x * 256 + threadIdx.x;
    if (i >= total4) return;
    long idx = (long)i * 4;
    int  w   = (int)(idx % 224);
    long t1  = idx / 224;
    int  hh  = (int)(t1 % 224);
    long t2  = t1 / 224;
    int  p   = (int)(t2 % 60);
    int  b   = (int)(t2 / 60);
    float v  = acts[(((size_t)(b * 60 + p)) * 14 + (hh >> 4)) * 14 + (w >> 4)];
    f32x4 r  = {v, v, v, v};
    __builtin_nontemporal_store(r, &out[i]);    // global_store_b128 ... nt
}

// ---------------------------------------------------------------------------
extern "C" void kernel_launch(void* const* d_in, const int* in_sizes, int n_in,
                              void* d_out, int out_size, void* d_ws, size_t ws_size,
                              hipStream_t stream) {
    const float* x      = (const float*)d_in[0];
    // d_in[1] mascaras: unused by the reference forward
    const float* Wf     = (const float*)d_in[2];
    const float* bf     = (const float*)d_in[3];
    const float* W1     = (const float*)d_in[4];
    const float* b1     = (const float*)d_in[5];
    const float* W2     = (const float*)d_in[6];
    const float* b2     = (const float*)d_in[7];
    const float* W3     = (const float*)d_in[8];
    const float* b3     = (const float*)d_in[9];
    const float* W4     = (const float*)d_in[10];
    const float* b4     = (const float*)d_in[11];
    const float* protos = (const float*)d_in[12];
    const float* lastW  = (const float*)d_in[13];
    const int*   topk   = (const int*)d_in[14];

    // ---- workspace carving (256B aligned bumps) ----
    char*  ws  = (char*)d_ws;
    size_t off = 0;
    auto carve = [&](size_t bytes) -> void* {
        void* pptr = ws + off;
        off += (bytes + 255) & ~(size_t)255;
        return pptr;
    };
    _Float16* Wf16  = (_Float16*)carve(393216 * sizeof(_Float16)); // 512x768
    _Float16* W1f16 = (_Float16*)carve(131072 * sizeof(_Float16)); // 256x512
    _Float16* W2f16 = (_Float16*)carve(65536  * sizeof(_Float16)); // 256x256
    _Float16* W3f16 = (_Float16*)carve(32768  * sizeof(_Float16)); // 128x256
    _Float16* W4f16 = (_Float16*)carve(16384  * sizeof(_Float16)); // 128x128
    _Float16* P     = (_Float16*)carve((size_t)3136 * 768 * sizeof(_Float16));
    _Float16* F0    = (_Float16*)carve((size_t)3136 * 512 * sizeof(_Float16));
    _Float16* H1    = (_Float16*)carve((size_t)3136 * 256 * sizeof(_Float16));
    _Float16* H2    = (_Float16*)carve((size_t)3136 * 256 * sizeof(_Float16));
    _Float16* H3    = (_Float16*)carve((size_t)3136 * 128 * sizeof(_Float16));
    _Float16* H4    = (_Float16*)carve((size_t)3136 * 128 * sizeof(_Float16));
    float*    H4f   = (float*)   carve((size_t)3136 * 128 * sizeof(float));
    float*    acts  = (float*)   carve((size_t)960 * 196 * sizeof(float));
    float*    pact  = (float*)   carve(960 * sizeof(float));

    float* out_logits = (float*)d_out;         // 16*3
    float* out_mind   = out_logits + 48;       // 16*60
    float* out_ups    = out_logits + 1008;     // 16*60*224*224 (16B aligned: 1008*4=4032)

    // ---- weight casts ----
    cast_f16_kernel<<<(393216 + 255) / 256, 256, 0, stream>>>(Wf, Wf16, 393216);
    cast_f16_kernel<<<(131072 + 255) / 256, 256, 0, stream>>>(W1, W1f16, 131072);
    cast_f16_kernel<<<(65536  + 255) / 256, 256, 0, stream>>>(W2, W2f16, 65536);
    cast_f16_kernel<<<(32768  + 255) / 256, 256, 0, stream>>>(W3, W3f16, 32768);
    cast_f16_kernel<<<(16384  + 255) / 256, 256, 0, stream>>>(W4, W4f16, 16384);

    // ---- im2col ----
    patchify_kernel<<<3136, 256, 0, stream>>>(x, P);

    // ---- WMMA GEMM chain (M = 3136 rows = 196 tile-rows, 16x32 per wave) ----
    auto gemm = [&](const _Float16* A, const _Float16* W, const float* bias,
                    _Float16* C, float* Cf32, int N, int K, int epi) {
        int tiles = 196 * (N >> 5);
        gemm_wmma_kernel<<<(tiles + 3) / 4, 128, 0, stream>>>(A, W, bias, C, Cf32,
                                                              3136, N, K, epi);
    };
    gemm(P,  Wf16, bf, F0, nullptr, 512, 768, 0);   // patch conv + ReLU
    gemm(F0, W1f16, b1, H1, nullptr, 256, 512, 0);  // 1x1 + ReLU
    gemm(H1, W2f16, b2, H2, nullptr, 256, 256, 0);  // 1x1 + ReLU
    gemm(H2, W3f16, b3, H3, nullptr, 128, 256, 0);  // 1x1 + ReLU
    gemm(H3, W4f16, b4, H4, H4f,    128, 128, 1);   // 1x1 + sigmoid (keep f32)

    // ---- distances, act maps, dual top-k ----
    dist_topk_kernel<<<960, 256, 0, stream>>>(H4f, protos, topk, acts, pact, out_mind);

    // ---- logits ----
    logits_kernel<<<1, 64, 0, stream>>>(pact, lastW, out_logits);

    // ---- 16x upsample (bandwidth-dominant: 193 MB of NT float4 stores) ----
    const int total4 = (16 * 60 * 224 * 224) / 4;   // 12,042,240
    upsample_kernel<<<(total4 + 255) / 256, 256, 0, stream>>>(acts, (f32x4*)out_ups, total4);
}
